// PE_GatedBlock_790273982669
// MI455X (gfx1250) — matrix operand order repaired and checked by reading, not compiled, not run on hardware
//
#include <hip/hip_runtime.h>
#include <hip/hip_bf16.h>
#include <math.h>

// MI455X / gfx1250, wave32. bf16x3 (hi/lo split) emulated-fp32 GEMMs on
// v_wmma_f32_16x16x32_bf16. A-matrices generated on the fly from LDS;
// weights pre-split into hi/lo bf16 ONCE per launch, stored in the WMMA
// B-operand register layout so each lane loads 16 contiguous-K bf16 values
// with two coalesced global_load_b128 (no per-chunk weight conversion).

typedef __attribute__((ext_vector_type(16))) __bf16 v16bf;
typedef __attribute__((ext_vector_type(8)))  float  v8f;

#define BTOT 8192
#define N0   128
#define N1   64
#define NS   128
#define NV   64
#define XW   320   // N0 + 3*N1
#define OUTW 320   // NS + 3*NV

#define K000 16384
#define K110 4096
#define K011 8192
#define K111 4096

#define WMMA_BF16(A, Bm, C) \
    __builtin_amdgcn_wmma_f32_16x16x32_bf16(false, (A), false, (Bm), (short)0, (C), false, false)

// Operand layouts used here (wave32; contiguous k order, not the ISA table order):
//   A 16x32 bf16 : lane L -> row m = L&15, half h = L>>4,
//                  elem e -> k_local = h*16 + e
//   B 32x16 bf16 : lane L -> col = L&15, elem e -> k = h*16 + e
//   C/D 16x16 f32: lane L -> col = L&15, elem r -> row = r + 8*h
// Packed weight layout: element (k, n) stored at ((k/32)*N + n)*32 + (k%32),
// so lane L reads its 16 values from base + ((kb/32)*N + col)*32 + h*16.

__device__ __forceinline__ void split2(float x, __bf16& hi, __bf16& lo) {
    hi = (__bf16)x;
    lo = (__bf16)(x - (float)hi);
}

// ---------------------------------------------------------------------------
// Weight prep: f32 (K x N) -> hi/lo bf16 in packed B-operand layout.
// ---------------------------------------------------------------------------
__global__ __launch_bounds__(256) void prep_split_kernel(
    const float* __restrict__ src, __bf16* __restrict__ dhi,
    __bf16* __restrict__ dlo, int total, int N)
{
    const int idx = blockIdx.x * 256 + threadIdx.x;
    if (idx >= total) return;
    const int k = idx / N, n = idx % N;
    __bf16 hi, lo;
    split2(src[idx], hi, lo);
    const size_t o = ((size_t)(k >> 5) * N + n) * 32 + (k & 31);
    dhi[o] = hi;  dlo[o] = lo;
}

// ---------------------------------------------------------------------------
// Scalars: one A-gen per chunk feeds 3 column tiles (ncol0, +64, +128).
// SEG 0: s000  A = x0[u]*x0[v] * (1/128),            K=16384
// SEG 1: s110  A = dot3(x1[u],x1[v]) * (1/sqrt3)/64, K=4096
// ---------------------------------------------------------------------------
template<int SEG>
__device__ __forceinline__ void gemm_sc(v8f* acc,
        const __bf16* __restrict__ Whi, const __bf16* __restrict__ Wlo, int K,
        const float* __restrict__ xs0, const float* __restrict__ xs1,
        int lane, int rowoff, int ncol0)
{
    const int m  = rowoff + (lane & 15);
    const int h  = lane >> 4;
    const int nn = lane & 15;

    for (int kb = 0; kb < K; kb += 32) {
        v16bf ahi, alo;
        if constexpr (SEG == 0) {
            const int u = kb >> 7, v0 = kb & 127;
            const float xu = xs0[m*128 + u] * 0.0078125f;
            #pragma unroll
            for (int e = 0; e < 16; ++e) {
                const int kl = h*16 + e;
                __bf16 hi, lo;
                split2(xu * xs0[m*128 + v0 + kl], hi, lo);
                ahi[e] = hi;  alo[e] = lo;
            }
        } else {
            const int u = kb >> 6, v0 = kb & 63;
            const float sc = 0.57735026919f / 64.0f;
            const float a0 = xs1[m*192 + u*3 + 0] * sc;
            const float a1 = xs1[m*192 + u*3 + 1] * sc;
            const float a2 = xs1[m*192 + u*3 + 2] * sc;
            #pragma unroll
            for (int e = 0; e < 16; ++e) {
                const int kl = h*16 + e;
                const float* bv = &xs1[m*192 + (v0 + kl)*3];
                __bf16 hi, lo;
                split2(a0*bv[0] + a1*bv[1] + a2*bv[2], hi, lo);
                ahi[e] = hi;  alo[e] = lo;
            }
        }
        const size_t cbase = (size_t)(kb >> 5) * (192 * 32) + (size_t)h * 16;
        #pragma unroll
        for (int t = 0; t < 3; ++t) {
            const size_t bo = cbase + (size_t)(ncol0 + t*64 + nn) * 32;
            const v16bf bhi = *(const v16bf*)(Whi + bo);
            const v16bf blo = *(const v16bf*)(Wlo + bo);
            acc[t] = WMMA_BF16(ahi, bhi, acc[t]);
            acc[t] = WMMA_BF16(ahi, blo, acc[t]);
            acc[t] = WMMA_BF16(alo, bhi, acc[t]);
        }
    }
}

// ---------------------------------------------------------------------------
// Vectors: one B-fetch per chunk feeds the 3 component accumulators.
// A values computed into f32 temps first, then packed per-vector so
// v_cvt_pk_bf16_f32 pairs adjacent elements (no v_mov_b16 shuffles).
// SEG 2: v011_i  A = x0[u]*x1[v,i] * (1/128),      K=8192
// SEG 3: v111_k  A = (x1[u] x x1[v])_k * (1/128),  K=4096
// ---------------------------------------------------------------------------
template<int SEG>
__device__ __forceinline__ void gemm_vec(v8f* acc,
        const __bf16* __restrict__ Whi, const __bf16* __restrict__ Wlo, int K,
        const float* __restrict__ xs0, const float* __restrict__ xs1,
        int lane, int rowoff, int ncol0)
{
    const int m  = rowoff + (lane & 15);
    const int h  = lane >> 4;
    const int nn = lane & 15;

    for (int kb = 0; kb < K; kb += 32) {
        const int u = kb >> 6, v0 = kb & 63;
        float cv[3][16];
        if constexpr (SEG == 2) {
            const float xu = xs0[m*128 + u] * 0.0078125f;
            #pragma unroll
            for (int e = 0; e < 16; ++e) {
                const int kl = h*16 + e;
                const float* bv = &xs1[m*192 + (v0 + kl)*3];
                cv[0][e] = xu * bv[0];
                cv[1][e] = xu * bv[1];
                cv[2][e] = xu * bv[2];
            }
        } else {
            const float a0 = xs1[m*192 + u*3 + 0] * 0.0078125f;
            const float a1 = xs1[m*192 + u*3 + 1] * 0.0078125f;
            const float a2 = xs1[m*192 + u*3 + 2] * 0.0078125f;
            #pragma unroll
            for (int e = 0; e < 16; ++e) {
                const int kl = h*16 + e;
                const float* bv = &xs1[m*192 + (v0 + kl)*3];
                cv[0][e] = a1*bv[2] - a2*bv[1];
                cv[1][e] = a2*bv[0] - a0*bv[2];
                cv[2][e] = a0*bv[1] - a1*bv[0];
            }
        }
        v16bf ahi[3], alo[3];
        #pragma unroll
        for (int ic = 0; ic < 3; ++ic) {
            #pragma unroll
            for (int e = 0; e < 16; ++e) {
                __bf16 hi, lo;
                split2(cv[ic][e], hi, lo);
                ahi[ic][e] = hi;  alo[ic][e] = lo;
            }
        }
        const size_t bo = (size_t)(kb >> 5) * (64 * 32) + (size_t)nn * 32 + (size_t)h * 16;
        const v16bf bhi = *(const v16bf*)(Whi + bo);
        const v16bf blo = *(const v16bf*)(Wlo + bo);
        #pragma unroll
        for (int ic = 0; ic < 3; ++ic) {
            acc[ic] = WMMA_BF16(ahi[ic], bhi, acc[ic]);
            acc[ic] = WMMA_BF16(ahi[ic], blo, acc[ic]);
            acc[ic] = WMMA_BF16(alo[ic], bhi, acc[ic]);
        }
    }
}

__device__ __forceinline__ void load_x_tile32(const float* __restrict__ x, int b0,
                                              int tid, float* xs0, float* xs1)
{
    for (int idx = tid; idx < 32*XW; idx += 256) {
        const int r = idx / XW, c = idx % XW;
        const float v = x[(size_t)(b0 + r) * XW + c];
        if (c < N0) xs0[r*128 + c] = v;
        else        xs1[r*192 + (c - N0)] = v;
    }
}

// ---------------------------------------------------------------------------
// Kernel 1: scalars = (s000 + s110)/sqrt2; cols<128 -> silu -> out,
//           cols>=128 -> sigmoid -> gate buffer. grid (B/32), block 256
//           = 2 row-subtiles x 4 column groups of 3 tiles.
// ---------------------------------------------------------------------------
__global__ __launch_bounds__(256) void pe_scalars_kernel(
    const float* __restrict__ x,
    const __bf16* __restrict__ w000hi, const __bf16* __restrict__ w000lo,
    const __bf16* __restrict__ w110hi, const __bf16* __restrict__ w110lo,
    float* __restrict__ out, float* __restrict__ gbuf)
{
    __shared__ float xs0[32*128];
    __shared__ float xs1[32*192];
    const int b0  = blockIdx.x * 32;
    const int tid = threadIdx.x;
    load_x_tile32(x, b0, tid, xs0, xs1);
    __syncthreads();

    const int lane   = tid & 31;
    const int wave   = tid >> 5;
    const int rowoff = (wave >> 2) * 16;
    const int ncol0  = (wave & 3) * 16;

    v8f acc[3] = {};
    gemm_sc<0>(acc, w000hi, w000lo, K000, xs0, xs1, lane, rowoff, ncol0);
    gemm_sc<1>(acc, w110hi, w110lo, K110, xs0, xs1, lane, rowoff, ncol0);

    const int h = lane >> 4;
    const int n = lane & 15;
    #pragma unroll
    for (int t = 0; t < 3; ++t) {
        const int col = ncol0 + t*64 + n;
        #pragma unroll
        for (int r = 0; r < 8; ++r) {
            const int   b  = b0 + rowoff + r + 8*h;
            const float s  = acc[t][r] * 0.70710678118f;        // * 1/sqrt2
            const float sg = 1.0f / (1.0f + __expf(-s));
            if (col < NS) out[(size_t)b * OUTW + col] = s * sg;  // silu
            else          gbuf[b * NV + (col - NS)]   = sg;      // gate
        }
    }
}

// ---------------------------------------------------------------------------
// Kernel 2: vecs_i = v011_i + v111_i; out[b, 128 + w*3 + i] = vecs_i * gate.
// grid (B/32), block 256 = 2 row-subtiles x 4 column tiles.
// ---------------------------------------------------------------------------
__global__ __launch_bounds__(256) void pe_vec_kernel(
    const float* __restrict__ x,
    const __bf16* __restrict__ w011hi, const __bf16* __restrict__ w011lo,
    const __bf16* __restrict__ w111hi, const __bf16* __restrict__ w111lo,
    const float* __restrict__ gbuf, float* __restrict__ out)
{
    __shared__ float xs0[32*128];
    __shared__ float xs1[32*192];
    const int b0  = blockIdx.x * 32;
    const int tid = threadIdx.x;
    load_x_tile32(x, b0, tid, xs0, xs1);
    __syncthreads();

    const int lane   = tid & 31;
    const int wave   = tid >> 5;
    const int rowoff = (wave >> 2) * 16;
    const int ncol0  = (wave & 3) * 16;

    v8f acc[3] = {};
    gemm_vec<2>(acc, w011hi, w011lo, K011, xs0, xs1, lane, rowoff, ncol0);
    gemm_vec<3>(acc, w111hi, w111lo, K111, xs0, xs1, lane, rowoff, ncol0);

    const int h = lane >> 4;
    const int w = ncol0 + (lane & 15);
    #pragma unroll
    for (int r = 0; r < 8; ++r) {
        const int   b = b0 + rowoff + r + 8*h;
        const float g = gbuf[b * NV + w];
        #pragma unroll
        for (int ic = 0; ic < 3; ++ic)
            out[(size_t)b * OUTW + NS + w*3 + ic] = acc[ic][r] * g;
    }
}

extern "C" void kernel_launch(void* const* d_in, const int* in_sizes, int n_in,
                              void* d_out, int out_size, void* d_ws, size_t ws_size,
                              hipStream_t stream)
{
    const float* x    = (const float*)d_in[0];
    const float* w000 = (const float*)d_in[1];
    const float* w110 = (const float*)d_in[2];
    const float* w011 = (const float*)d_in[3];
    const float* w111 = (const float*)d_in[4];
    float* out  = (float*)d_out;

    // Workspace layout: [gate buffer f32 B*64][8 packed bf16 weight arrays]
    float*  gbuf = (float*)d_ws;
    __bf16* wp   = (__bf16*)((char*)d_ws + (size_t)BTOT * NV * sizeof(float));
    size_t o = 0;
    __bf16* w000hi = wp + o;  o += (size_t)K000 * 192;
    __bf16* w000lo = wp + o;  o += (size_t)K000 * 192;
    __bf16* w110hi = wp + o;  o += (size_t)K110 * 192;
    __bf16* w110lo = wp + o;  o += (size_t)K110 * 192;
    __bf16* w011hi = wp + o;  o += (size_t)K011 * 64;
    __bf16* w011lo = wp + o;  o += (size_t)K011 * 64;
    __bf16* w111hi = wp + o;  o += (size_t)K111 * 64;
    __bf16* w111lo = wp + o;  o += (size_t)K111 * 64;

    // One-pass weight split (~19 MB, bandwidth-trivial; runs every call).
    auto prep = [&](const float* s, __bf16* hi, __bf16* lo, int K, int N) {
        const int tot = K * N;
        prep_split_kernel<<<(tot + 255) / 256, 256, 0, stream>>>(s, hi, lo, tot, N);
    };
    prep(w000, w000hi, w000lo, K000, 192);
    prep(w110, w110hi, w110lo, K110, 192);
    prep(w011, w011hi, w011lo, K011, 64);
    prep(w111, w111hi, w111lo, K111, 64);

    pe_scalars_kernel<<<dim3(BTOT/32), 256, 0, stream>>>(
        x, w000hi, w000lo, w110hi, w110lo, out, gbuf);
    pe_vec_kernel<<<dim3(BTOT/32), 256, 0, stream>>>(
        x, w011hi, w011lo, w111hi, w111lo, gbuf, out);
}
